// MultiHeadAttention_8108898255307
// MI455X (gfx1250) — matrix operand order repaired and checked by reading, not compiled, not run on hardware
//
#include <hip/hip_runtime.h>

// ---------------------------------------------------------------------------
// MI455X / gfx1250 multi-head attention with relative position bias.
// Strategy: cast everything to bf16, do all matmuls with
// v_wmma_f32_16x16x32_bf16 (f32 accumulate), flash-attention with the
// S^T = K*Q^T trick so softmax rows live inside a lane (wave32 layouts).
// ---------------------------------------------------------------------------

typedef __attribute__((ext_vector_type(16))) __bf16   v16bf;
typedef __attribute__((ext_vector_type(8)))  float    v8f;
typedef __attribute__((ext_vector_type(4)))  unsigned uint4v;
typedef __attribute__((ext_vector_type(4)))  int      int4v;
typedef unsigned short u16;

constexpr int Dm  = 1024;   // d_model
constexpr int Sq  = 2048;   // sequence length
constexpr int NH  = 16;     // heads
constexpr int DKh = 64;     // head dim
constexpr int Bb  = 4;      // batch
constexpr int Mrows = Bb * Sq;  // 8192

union Frag16 {              // one 16-bf16 WMMA operand (8 VGPRs)
    v16bf  v;
    uint4v q[2];
    unsigned u[8];
};

static __device__ __forceinline__ unsigned rne_bf16(float f) {
    unsigned u = __builtin_bit_cast(unsigned, f);
    return (u + 0x7FFFu + ((u >> 16) & 1u)) >> 16;   // round-to-nearest-even
}
static __device__ __forceinline__ unsigned pk_bf16(float lo, float hi) {
    return (rne_bf16(lo) & 0xFFFFu) | (rne_bf16(hi) << 16);
}

// ---------------------------------------------------------------------------
// f32 -> bf16 elementwise conversion (bandwidth trivial vs 23.3 TB/s)
// ---------------------------------------------------------------------------
__global__ void cvt_f32_bf16(const float* __restrict__ x, u16* __restrict__ y, int n) {
    int i = blockIdx.x * blockDim.x + threadIdx.x;
    int stride = gridDim.x * blockDim.x;
    for (; i < n; i += stride) y[i] = (u16)rne_bf16(x[i]);
}

// ---------------------------------------------------------------------------
// WMMA GEMM: Y = X[M,1024] @ W[N,1024]^T + bias.  64x64 block tile,
// 32x32 per wave (4 waves), K-step 32.  Fragments loaded straight from
// global; 192MB L2 keeps X and W hot across the grid.
// MODE 0: store bf16 [B,H,S,dk]      (Q and K projections)
// MODE 2: store bf16 [B,H,dk,S]      (V projection, pre-transposed for PV)
// MODE 3: store f32  flat [M,1024]   (output projection)
// ---------------------------------------------------------------------------
template<int MODE>
__global__ __launch_bounds__(128, 2)
void gemm_wmma(const u16* __restrict__ X, const u16* __restrict__ W,
               const float* __restrict__ bias, void* __restrict__ Y) {
    const int lane = threadIdx.x & 31;
    const int wv   = threadIdx.x >> 5;
    const int hl   = lane >> 4;      // lane half (0/1)
    const int ln   = lane & 15;
    const int m0 = blockIdx.x * 64 + (wv >> 1) * 32;
    const int n0 = blockIdx.y * 64 + (wv & 1) * 32;

    v8f c[2][2] = {};
    for (int k0 = 0; k0 < Dm; k0 += 32) {
        Frag16 a[2], bfr[2];
#pragma unroll
        for (int mi = 0; mi < 2; ++mi) {
            // A 16x32 bf16 layout: lane ln holds row m; contiguous K run of 16 at 16*hl
            const u16* pa = X + (size_t)(m0 + mi * 16 + ln) * Dm + k0 + 16 * hl;
            a[mi].q[0] = *(const uint4v*)pa;
            a[mi].q[1] = *(const uint4v*)(pa + 8);
        }
#pragma unroll
        for (int ni = 0; ni < 2; ++ni) {
            // B 32x16 bf16 layout: lane ln holds col n; contiguous K run of 16 at 16*hl
            const u16* pb = W + (size_t)(n0 + ni * 16 + ln) * Dm + k0 + 16 * hl;
            bfr[ni].q[0] = *(const uint4v*)pb;
            bfr[ni].q[1] = *(const uint4v*)(pb + 8);
        }
        if (k0 + 32 < Dm) {  // gfx1250 global_prefetch path for next K tile
            __builtin_prefetch((const void*)(X + (size_t)(m0 + ln) * Dm + k0 + 32), 0, 1);
        }
#pragma unroll
        for (int mi = 0; mi < 2; ++mi)
#pragma unroll
            for (int ni = 0; ni < 2; ++ni)
                c[mi][ni] = __builtin_amdgcn_wmma_f32_16x16x32_bf16(
                    false, a[mi].v, false, bfr[ni].v, (short)0, c[mi][ni], false, false);
    }

#pragma unroll
    for (int mi = 0; mi < 2; ++mi)
#pragma unroll
        for (int ni = 0; ni < 2; ++ni) {
            const int n  = n0 + ni * 16 + ln;           // C layout: lane holds column n
            const float bv = bias[n];
#pragma unroll
            for (int r = 0; r < 8; ++r) {
                const int m = m0 + mi * 16 + r + 8 * hl; // row = VGPR r + 8*half
                const float val = c[mi][ni][r] + bv;
                if (MODE == 3) {
                    ((float*)Y)[(size_t)m * Dm + n] = val;
                } else {
                    const int bi = m >> 11, s = m & (Sq - 1);
                    const int hh = n >> 6,  dkk = n & (DKh - 1);
                    size_t idx;
                    if (MODE == 0) idx = (((size_t)bi * NH + hh) * Sq + s) * DKh + dkk;
                    else           idx = (((size_t)bi * NH + hh) * DKh + dkk) * Sq + s;
                    ((u16*)Y)[idx] = (u16)rne_bf16(val);
                }
            }
        }
}

// ---------------------------------------------------------------------------
// Flash attention, one 16-query tile per wave, 4 waves/block (same b,h).
// Computes S^T = K * Q^T so each lane owns one query column:
//   S^T C-layout: lane l -> (kv = r + 8*(l>=16), q = l&15)  => softmax
//   reductions are in-lane over 8 VGPRs + one shfl_xor(16).
// Then O^T = V^T * P^T:  P^T rebuilt as a B-fragment with one cross-half
// shuffle exchange; V was stored transposed so V^T A-fragments load
// contiguously.  Rel-emb column staged in LDS.
// ---------------------------------------------------------------------------
__global__ __launch_bounds__(128, 1)
void attn_kernel(const u16* __restrict__ Q, const u16* __restrict__ K,
                 const u16* __restrict__ V, const int* __restrict__ mask,
                 const float* __restrict__ rel, u16* __restrict__ ctx) {
    __shared__ float relf[257];
    const int tid  = threadIdx.x;
    const int wv   = tid >> 5;
    const int lane = tid & 31;
    const int hl   = lane >> 4;
    const int ln   = lane & 15;

    // all 4 tiles in a block share (b,h): 128 q-tiles per head, 4 | 128
    const int hblk = ((blockIdx.x * 4) >> 7) & (NH - 1);
    for (int i = tid; i < 257; i += 128) relf[i] = rel[(size_t)i * NH + hblk];
    __syncthreads();

    const int t  = blockIdx.x * 4 + wv;
    const int b  = t >> 11;
    const int h  = (t >> 7) & (NH - 1);
    const int qt = t & 127;
    const int qrow = qt * 16 + ln;

    const u16* Qb = Q + (((size_t)b * NH + h) * Sq + qrow) * DKh;
    const u16* Kb = K + ((size_t)b * NH + h) * Sq * DKh;
    const u16* Vb = V + ((size_t)b * NH + h) * DKh * Sq;
    const int* mb = mask + b * Sq;

    // Q^T as B-fragments (lane = q column, contiguous dk run of 16), loaded once
    Frag16 qf[2];
#pragma unroll
    for (int kk = 0; kk < 2; ++kk) {
        const u16* p = Qb + kk * 32 + 16 * hl;
        qf[kk].q[0] = *(const uint4v*)p;
        qf[kk].q[1] = *(const uint4v*)(p + 8);
    }

    float Mx = -1e30f, L = 0.f;
    v8f o[4] = {};   // O^T accumulators, 4 dk-tiles of 16

    for (int kt = 0; kt < Sq; kt += 32) {
        float sc[2][8];
#pragma unroll
        for (int sub = 0; sub < 2; ++sub) {
            const int kv0 = kt + sub * 16;
            Frag16 ka[2];
            const u16* Kr = Kb + (size_t)(kv0 + ln) * DKh;
#pragma unroll
            for (int kk = 0; kk < 2; ++kk) {   // A-frag: contiguous run of 16 at 16*hl
                const u16* p = Kr + kk * 32 + 16 * hl;
                ka[kk].q[0] = *(const uint4v*)p;
                ka[kk].q[1] = *(const uint4v*)(p + 8);
            }
            v8f cc = {};
            cc = __builtin_amdgcn_wmma_f32_16x16x32_bf16(false, ka[0].v, false, qf[0].v,
                                                         (short)0, cc, false, false);
            cc = __builtin_amdgcn_wmma_f32_16x16x32_bf16(false, ka[1].v, false, qf[1].v,
                                                         (short)0, cc, false, false);
            union { int4v v[2]; int a[8]; } mk;
            const int* mp = mb + kv0 + 8 * hl;
            mk.v[0] = *(const int4v*)mp;
            mk.v[1] = *(const int4v*)(mp + 4);
#pragma unroll
            for (int r = 0; r < 8; ++r) {
                const int kv = kv0 + 8 * hl + r;
                int dd = kv - qrow;
                dd = dd < -128 ? -128 : (dd > 128 ? 128 : dd);
                const float sv = cc[r] * 0.125f + relf[dd + 128];
                sc[sub][r] = mk.a[r] ? sv : -1e30f;
            }
        }
        // online softmax: per-lane max over 16 values + cross-half combine
        float tm = sc[0][0];
#pragma unroll
        for (int r = 1; r < 8; ++r) tm = fmaxf(tm, sc[0][r]);
#pragma unroll
        for (int r = 0; r < 8; ++r) tm = fmaxf(tm, sc[1][r]);
        tm = fmaxf(tm, __shfl_xor(tm, 16, 32));
        const float Mn    = fmaxf(Mx, tm);
        const float alpha = __expf(Mx - Mn);
#pragma unroll
        for (int d = 0; d < 4; ++d) o[d] *= alpha;
        L *= alpha;
        float p0[8], p1[8], ps = 0.f;
#pragma unroll
        for (int r = 0; r < 8; ++r) {
            p0[r] = __expf(sc[0][r] - Mn);
            p1[r] = __expf(sc[1][r] - Mn);
            ps += p0[r] + p1[r];
        }
        L += ps;
        Mx = Mn;
        // build P^T B-fragment (32 kv x 16 q): cross-half exchange + bf16 pack
        float xp0[8], xp1[8];
#pragma unroll
        for (int r = 0; r < 8; ++r) {
            xp0[r] = __shfl_xor(p0[r], 16, 32);
            xp1[r] = __shfl_xor(p1[r], 16, 32);
        }
        Frag16 pf;
#pragma unroll
        for (int i = 0; i < 4; ++i) {
            const float a0 = hl ? xp1[2 * i]     : p0[2 * i];
            const float a1 = hl ? xp1[2 * i + 1] : p0[2 * i + 1];
            const float b0 = hl ? p1[2 * i]      : xp0[2 * i];
            const float b1 = hl ? p1[2 * i + 1]  : xp0[2 * i + 1];
            pf.u[i]     = pk_bf16(a0, a1);
            pf.u[4 + i] = pk_bf16(b0, b1);
        }
        // O^T += V^T * P^T   (V stored [dk][s] so A-frags are contiguous)
#pragma unroll
        for (int d = 0; d < 4; ++d) {
            Frag16 va;
            const u16* p = Vb + (size_t)(d * 16 + ln) * Sq + kt + 16 * hl;
            va.q[0] = *(const uint4v*)p;
            va.q[1] = *(const uint4v*)(p + 8);
            o[d] = __builtin_amdgcn_wmma_f32_16x16x32_bf16(false, va.v, false, pf.v,
                                                           (short)0, o[d], false, false);
        }
    }
    L += __shfl_xor(L, 16, 32);
    const float inv = (Mx > -1e29f && L > 0.f) ? 1.f / L : 0.f; // fully-masked -> 0
    u16* cb = ctx + ((size_t)b * Sq + qrow) * Dm + h * DKh;      // merge heads
#pragma unroll
    for (int d = 0; d < 4; ++d)
#pragma unroll
        for (int r = 0; r < 8; ++r)
            cb[d * 16 + r + 8 * hl] = (u16)rne_bf16(o[d][r] * inv);
}

// ---------------------------------------------------------------------------
extern "C" void kernel_launch(void* const* d_in, const int* in_sizes, int n_in,
                              void* d_out, int out_size, void* d_ws, size_t ws_size,
                              hipStream_t stream) {
    const float* query = (const float*)d_in[0];
    const float* key_  = (const float*)d_in[1];
    const float* value = (const float*)d_in[2];
    const int*   mask  = (const int*)d_in[3];
    const float* Wq = (const float*)d_in[4];
    const float* bq = (const float*)d_in[5];
    const float* Wk = (const float*)d_in[6];
    const float* bk = (const float*)d_in[7];
    const float* Wv = (const float*)d_in[8];
    const float* bv = (const float*)d_in[9];
    const float* Wo = (const float*)d_in[10];
    const float* bo = (const float*)d_in[11];
    const float* rel = (const float*)d_in[12];

    const size_t ACT = (size_t)Mrows * Dm * sizeof(u16);   // 16 MiB
    const size_t WSZ = (size_t)Dm * Dm * sizeof(u16);      //  2 MiB
    char* w = (char*)d_ws;
    u16* qb  = (u16*)w; w += ACT;
    u16* kb  = (u16*)w; w += ACT;
    u16* vb  = (u16*)w; w += ACT;
    u16* wqb = (u16*)w; w += WSZ;
    u16* wkb = (u16*)w; w += WSZ;
    u16* wvb = (u16*)w; w += WSZ;
    u16* wob = (u16*)w; w += WSZ;
    u16* Qbh = (u16*)w; w += ACT;
    u16* Kbh = (u16*)w; w += ACT;
    u16* Vtb = (u16*)w; w += ACT;
    u16* ctx = (u16*)w; w += ACT;
    (void)in_sizes; (void)n_in; (void)out_size; (void)ws_size;

    const int nAct = Mrows * Dm;    // 8388608
    const int nW   = Dm * Dm;       // 1048576
    cvt_f32_bf16<<<2048, 256, 0, stream>>>(query, qb, nAct);
    cvt_f32_bf16<<<2048, 256, 0, stream>>>(key_,  kb, nAct);
    cvt_f32_bf16<<<2048, 256, 0, stream>>>(value, vb, nAct);
    cvt_f32_bf16<<<1024, 256, 0, stream>>>(Wq, wqb, nW);
    cvt_f32_bf16<<<1024, 256, 0, stream>>>(Wk, wkb, nW);
    cvt_f32_bf16<<<1024, 256, 0, stream>>>(Wv, wvb, nW);
    cvt_f32_bf16<<<1024, 256, 0, stream>>>(Wo, wob, nW);

    const dim3 ggrid(Mrows / 64, Dm / 64);   // 128 x 16
    gemm_wmma<0><<<ggrid, 128, 0, stream>>>(qb, wqb, bq, (void*)Qbh);
    gemm_wmma<0><<<ggrid, 128, 0, stream>>>(kb, wkb, bk, (void*)Kbh);
    gemm_wmma<2><<<ggrid, 128, 0, stream>>>(vb, wvb, bv, (void*)Vtb);

    // B*H*(S/16) = 8192 q-tiles, 4 per block
    attn_kernel<<<2048, 128, 0, stream>>>(Qbh, Kbh, Vtb, mask, rel, ctx);

    gemm_wmma<3><<<ggrid, 128, 0, stream>>>(ctx, wob, bo, d_out);
}
